// ClassSeparationLossMargin_15504831938685
// MI455X (gfx1250) — matrix operand order repaired and checked by reading, not compiled, not run on hardware
//
#include <hip/hip_runtime.h>
#include <hip/hip_bf16.h>

// ---------------------------------------------------------------------------
// ClassSeparationLossMargin, fused for MI455X (gfx1250, wave32, WMMA).
//
// loss = mean over the 8192x8192 cosine-similarity Gram matrix of
//          t*dist + (1-t)*relu(MARGIN - dist),  dist = 1 - cos
// Never materializes the N x N matrix: each wave computes 16x16 f32 tiles
// with v_wmma_f32_16x16x32_f16 and folds the loss + reduction in registers.
// ---------------------------------------------------------------------------

typedef __attribute__((ext_vector_type(16))) _Float16 v16h;
typedef __attribute__((ext_vector_type(8)))  _Float16 v8h;
typedef __attribute__((ext_vector_type(8)))  float    v8f;

#define CSL_N        8192
#define CSL_D        64
#define CSL_NTILES   (CSL_N / 16)          // 512 tiles per dimension
#define CSL_JCHUNKS  32                    // 16 j-tiles per wave
#define CSL_WAVES    (CSL_NTILES * CSL_JCHUNKS)   // 16384 waves
#define CSL_MARGIN   1.1f
#define CSL_EPS      1e-8f

// --------------------------- kernel 1: normalize ---------------------------
// f32 [N,D] -> unit-normalized f16 [N,D] in workspace. 2 MB read, 1 MB write.
__global__ void __launch_bounds__(256)
csl_normalize_rows(const float* __restrict__ x, _Float16* __restrict__ xn)
{
    const int row = blockIdx.x * blockDim.x + threadIdx.x;
    if (row >= CSL_N) return;

    const float4* xr = (const float4*)(x + row * CSL_D);
    float4 v[CSL_D / 4];
    float s = 0.f;
#pragma unroll
    for (int i = 0; i < CSL_D / 4; ++i) {
        v[i] = xr[i];
        s += v[i].x * v[i].x + v[i].y * v[i].y + v[i].z * v[i].z + v[i].w * v[i].w;
    }
    const float inv = 1.f / fmaxf(sqrtf(s), CSL_EPS);

    v8h* out = (v8h*)(xn + row * CSL_D);
#pragma unroll
    for (int i = 0; i < CSL_D / 8; ++i) {
        const float4 a = v[2 * i], b = v[2 * i + 1];
        v8h h;
        h[0] = (_Float16)(a.x * inv); h[1] = (_Float16)(a.y * inv);
        h[2] = (_Float16)(a.z * inv); h[3] = (_Float16)(a.w * inv);
        h[4] = (_Float16)(b.x * inv); h[5] = (_Float16)(b.y * inv);
        h[6] = (_Float16)(b.z * inv); h[7] = (_Float16)(b.w * inv);
        out[i] = h;
    }
}

// ------------------- kernel 2: fused Gram x loss x reduce -------------------
// One wave per (i-tile, j-chunk). A fragments loaded once per wave; 16 j-tiles
// each need 2 B fragments + 2 WMMAs + an 8-element VALU epilogue per lane.
__global__ void __launch_bounds__(256)
csl_gram_loss(const _Float16* __restrict__ xn, const int* __restrict__ cls,
              float* __restrict__ partials)
{
    const int lane   = threadIdx.x & 31;
    const int wave   = blockIdx.x * (blockDim.x >> 5) + (threadIdx.x >> 5);
    const int iTile  = wave >> 5;           // 0..511
    const int jChunk = wave & (CSL_JCHUNKS - 1);
    const int i0     = iTile * 16;
    const int hi     = lane >> 4;           // lane half (0/1)
    const int lo     = lane & 15;

    // ---- A fragments (16x32 f16 each), contiguous k order (same as B):
    // lanes 0-15 : M=lane,    halves 0-15 = K 0-15
    // lanes 16-31: M=lane-16, halves 0-15 = K 16-31
    v16h afrag[2];
    {
        const _Float16* rowp = xn + (size_t)(i0 + lo) * CSL_D;
#pragma unroll
        for (int f = 0; f < 2; ++f) {
            const _Float16* p  = rowp + f * 32 + hi * 16;
            const v8h lo8 = *(const v8h*)(p);
            const v8h hi8 = *(const v8h*)(p + 8);
            v16h a;
#pragma unroll
            for (int e = 0; e < 8; ++e) { a[e] = lo8[e]; a[8 + e] = hi8[e]; }
            afrag[f] = a;
        }
    }

    // C-layout row classes: element v in lane => M = v + 8*hi
    int cmI[8];
#pragma unroll
    for (int v = 0; v < 8; ++v) cmI[v] = cls[i0 + v + 8 * hi];

    float acc = 0.f;
    const int jBase = jChunk * 16 * 16;     // first j row of this chunk

    for (int jt = 0; jt < 16; ++jt) {
        const int j0 = jBase + jt * 16;

        // ---- B fragments (32x16 f16), contiguous k order (same as A):
        // lanes 0-15 : N=lane,    halves 0-15 = K kBase..kBase+15
        // lanes 16-31: N=lane-16, halves 0-15 = K kBase+16..kBase+31
        // Column N of B is row (j0+N) of xn => per-lane contiguous 32 bytes.
        const _Float16* bp = xn + (size_t)(j0 + lo) * CSL_D + hi * 16;
        const v16h b0 = *(const v16h*)(bp);        // K = 0..31
        const v16h b1 = *(const v16h*)(bp + 32);   // K = 32..63

        v8f c = {};
        c = __builtin_amdgcn_wmma_f32_16x16x32_f16(false, afrag[0], false, b0,
                                                   (short)0, c, false, false);
        c = __builtin_amdgcn_wmma_f32_16x16x32_f16(false, afrag[1], false, b1,
                                                   (short)0, c, false, false);

        const int cj = cls[j0 + lo];               // column class (N = lane&15)
#pragma unroll
        for (int v = 0; v < 8; ++v) {
            const float dist = 1.f - c[v];
            const float neg  = fmaxf(CSL_MARGIN - dist, 0.f);
            acc += (cj == cmI[v]) ? dist : neg;
        }
    }

    // wave32 reduction
#pragma unroll
    for (int off = 16; off > 0; off >>= 1)
        acc += __shfl_down(acc, off, 32);
    if (lane == 0) partials[wave] = acc;
}

// ------------------------ kernel 3: final reduction ------------------------
__global__ void __launch_bounds__(256)
csl_reduce_partials(const float* __restrict__ partials, float* __restrict__ out)
{
    __shared__ float sm[256];
    float s = 0.f;
    for (int i = threadIdx.x; i < CSL_WAVES; i += 256) s += partials[i];
    sm[threadIdx.x] = s;
    __syncthreads();
#pragma unroll
    for (int off = 128; off > 0; off >>= 1) {
        if (threadIdx.x < off) sm[threadIdx.x] += sm[threadIdx.x + off];
        __syncthreads();
    }
    if (threadIdx.x == 0)
        out[0] = sm[0] * (1.f / ((float)CSL_N * (float)CSL_N));
}

// --------------------------------- launch ----------------------------------
extern "C" void kernel_launch(void* const* d_in, const int* in_sizes, int n_in,
                              void* d_out, int out_size, void* d_ws, size_t ws_size,
                              hipStream_t stream)
{
    (void)in_sizes; (void)n_in; (void)out_size; (void)ws_size;

    const float* bottleneck = (const float*)d_in[0];   // [8192,64] f32
    const int*   class_map  = (const int*)d_in[1];     // [8192]    i32
    float*       out        = (float*)d_out;           // scalar f32

    // Workspace layout: xn f16 (1 MiB) | per-wave partials (64 KiB)
    _Float16* xn       = (_Float16*)d_ws;
    float*    partials = (float*)((char*)d_ws + (size_t)CSL_N * CSL_D * sizeof(_Float16));

    csl_normalize_rows<<<CSL_N / 256, 256, 0, stream>>>(bottleneck, xn);
    csl_gram_loss<<<CSL_WAVES / 8, 256, 0, stream>>>(xn, class_map, partials);
    csl_reduce_partials<<<1, 256, 0, stream>>>(partials, out);
}
